// MyLSTM_17102559773335
// MI455X (gfx1250) — compile-verified, matrix operand order checked by reading, not run on hardware
//
#include <hip/hip_runtime.h>
#include <hip/hip_bf16.h>

// ---------------------------------------------------------------------------
// Types for CDNA5 WMMA
// ---------------------------------------------------------------------------
typedef __attribute__((ext_vector_type(16))) __bf16        v16bf;
typedef __attribute__((ext_vector_type(2)))  __bf16        v2bf;
typedef __attribute__((ext_vector_type(8)))  float         v8f;
typedef __attribute__((ext_vector_type(8)))  unsigned short us8;

// f32 -> bf16 via hardware converter (RNE): lowers to v_cvt_pk_bf16_f32.
__device__ __forceinline__ unsigned int pack2bf(float lo, float hi) {
    v2bf v;
    v.x = (__bf16)lo;
    v.y = (__bf16)hi;
    return __builtin_bit_cast(unsigned int, v);
}

__device__ __forceinline__ float sigm(float x) {
    return 1.0f / (1.0f + __expf(-x));
}

// ---------------------------------------------------------------------------
// Kernel 1: C[M,N] = A[M,K] @ B[K,N], f32 in/out, bf16 WMMA accumulate-f32.
// Block = 256 threads (8 waves), 128x128 tile, BK = 32.
// Software pipelined with ping-pong register sets (loop unrolled x2 chunks).
// ---------------------------------------------------------------------------
#define BM 128
#define BN 128
#define BK 32
#define SA 40   // LDS row stride (ushorts): 80B/row -> 16B-aligned frag reads
#define SB 40

// Per-thread chunk share: A = 4 float4 (row-quads), B = 2 k-pairs of float4.
template <bool GUARD>
__device__ __forceinline__ void load_chunk(const float* __restrict__ A,
                                           const float* __restrict__ B,
                                           int m0, int n0, int K, int N, int k0,
                                           int tid, float4* ra,
                                           float4* rbl, float4* rbh) {
    #pragma unroll
    for (int i = 0; i < 4; ++i) {
        int idx = tid + i * 256;               // 0..1023
        int row = idx >> 3;                    // 0..127
        int k   = (idx & 7) << 2;              // 0..28 step 4 (K % 4 == 0)
        if (!GUARD || (k0 + k < K))
            ra[i] = *(const float4*)(A + (size_t)(m0 + row) * K + k0 + k);
        else
            ra[i] = make_float4(0.f, 0.f, 0.f, 0.f);
    }
    #pragma unroll
    for (int i = 0; i < 2; ++i) {
        int p  = tid + i * 256;                // 0..511
        int kp = p >> 5;                       // k-pair 0..15
        int n  = (p & 31) << 2;                // 0..124 step 4
        int k  = k0 + 2 * kp;
        if (!GUARD || (k < K))                 // K even -> k valid => pair row0 ok
            rbl[i] = *(const float4*)(B + (size_t)k * N + n0 + n);
        else
            rbl[i] = make_float4(0.f, 0.f, 0.f, 0.f);
        if (!GUARD || (k + 1 < K))
            rbh[i] = *(const float4*)(B + (size_t)(k + 1) * N + n0 + n);
        else
            rbh[i] = make_float4(0.f, 0.f, 0.f, 0.f);
    }
}

// Convert + store one chunk's registers into the LDS tiles.
__device__ __forceinline__ void stage_chunk(unsigned short* __restrict__ As,
                                            unsigned short* __restrict__ Bs,
                                            int tid, const float4* ra,
                                            const float4* rbl, const float4* rbh) {
    #pragma unroll
    for (int i = 0; i < 4; ++i) {
        int idx = tid + i * 256;
        int row = idx >> 3;
        int k   = (idx & 7) << 2;
        uint2 p = make_uint2(pack2bf(ra[i].x, ra[i].y),
                             pack2bf(ra[i].z, ra[i].w));
        *(uint2*)&As[row * SA + k] = p;        // 8B-aligned
    }
    #pragma unroll
    for (int i = 0; i < 2; ++i) {
        int p  = tid + i * 256;
        int kp = p >> 5;
        int n  = (p & 31) << 2;
        int kk = 2 * kp;
        // one packed convert + one b32 store per n-row of the k-pair
        *(unsigned int*)&Bs[(n + 0) * SB + kk] = pack2bf(rbl[i].x, rbh[i].x);
        *(unsigned int*)&Bs[(n + 1) * SB + kk] = pack2bf(rbl[i].y, rbh[i].y);
        *(unsigned int*)&Bs[(n + 2) * SB + kk] = pack2bf(rbl[i].z, rbh[i].z);
        *(unsigned int*)&Bs[(n + 3) * SB + kk] = pack2bf(rbl[i].w, rbh[i].w);
    }
}

// Fragment loads + 8 WMMAs on the current LDS tiles.
__device__ __forceinline__ void compute_chunk(const unsigned short* __restrict__ As,
                                              const unsigned short* __restrict__ Bs,
                                              int wm, int wn, int half, int l16,
                                              v8f acc[2][4]) {
    // 16-bit A 16x32 layout: elems 0..7 -> k = half*8 + e,
    // elems 8..15 -> k = 16 + half*8 + (e-8): two contiguous 16B runs.
    union Frag { v16bf v; us8 u[2]; };
    Frag fa[2], fb[4];
    #pragma unroll
    for (int tm = 0; tm < 2; ++tm) {
        int m = wm * 32 + tm * 16 + l16;
        fa[tm].u[0] = *(const us8*)&As[m * SA + half * 8];
        fa[tm].u[1] = *(const us8*)&As[m * SA + 16 + half * 8];
    }
    #pragma unroll
    for (int tn = 0; tn < 4; ++tn) {
        int n = wn * 64 + tn * 16 + l16;
        fb[tn].u[0] = *(const us8*)&Bs[n * SB + half * 8];
        fb[tn].u[1] = *(const us8*)&Bs[n * SB + 16 + half * 8];
    }
    #pragma unroll
    for (int tm = 0; tm < 2; ++tm)
        #pragma unroll
        for (int tn = 0; tn < 4; ++tn)
            acc[tm][tn] = __builtin_amdgcn_wmma_f32_16x16x32_bf16(
                false, fa[tm].v, false, fb[tn].v,
                (short)0, acc[tm][tn], false, false);
}

__global__ void __launch_bounds__(256)
gemm_bf16_wmma(const float* __restrict__ A, const float* __restrict__ B,
               float* __restrict__ C, int M, int N, int K) {
    __shared__ unsigned short As[BM * SA];   // A tile, [m][k], bf16
    __shared__ unsigned short Bs[BN * SB];   // B tile transposed, [n][k], bf16

    const int tid  = threadIdx.x;
    const int lane = tid & 31;
    const int w    = tid >> 5;
    const int wm   = w & 3;        // 4 M-groups of 32 rows
    const int wn   = w >> 2;       // 2 N-groups of 64 cols
    const int half = lane >> 4;    // K-half select per WMMA layout
    const int l16  = lane & 15;

    const int m0 = blockIdx.x * BM;
    const int n0 = blockIdx.y * BN;

    v8f acc[2][4];
    #pragma unroll
    for (int i = 0; i < 2; ++i)
        #pragma unroll
        for (int j = 0; j < 4; ++j) acc[i][j] = {};

    float4 ra[4], rbl[2], rbh[2];      // ping
    float4 ra2[4], rbl2[2], rbh2[2];   // pong

    // Prologue: fetch chunk 0.
    if (BK <= K) load_chunk<false>(A, B, m0, n0, K, N, 0, tid, ra, rbl, rbh);
    else         load_chunk<true >(A, B, m0, n0, K, N, 0, tid, ra, rbl, rbh);

    for (int k0 = 0; k0 < K; k0 += 2 * BK) {
        // ---------------- even chunk (ping registers) ----------------
        stage_chunk(As, Bs, tid, ra, rbl, rbh);
        __syncthreads();
        {
            const int kn = k0 + BK;
            if (kn + BK <= K)
                load_chunk<false>(A, B, m0, n0, K, N, kn, tid, ra2, rbl2, rbh2);
            else if (kn < K)
                load_chunk<true >(A, B, m0, n0, K, N, kn, tid, ra2, rbl2, rbh2);
            if (kn + BK < K) {   // warm GL2 two chunks ahead
                __builtin_prefetch(A + (size_t)(m0 + (tid >> 3)) * K + kn + BK, 0, 1);
                __builtin_prefetch(B + (size_t)(kn + BK + (tid >> 5)) * N + n0, 0, 1);
            }
        }
        compute_chunk(As, Bs, wm, wn, half, l16, acc);
        __syncthreads();
        if (k0 + BK >= K) break;

        // ---------------- odd chunk (pong registers) ----------------
        stage_chunk(As, Bs, tid, ra2, rbl2, rbh2);
        __syncthreads();
        {
            const int kn = k0 + 2 * BK;
            if (kn + BK <= K)
                load_chunk<false>(A, B, m0, n0, K, N, kn, tid, ra, rbl, rbh);
            else if (kn < K)
                load_chunk<true >(A, B, m0, n0, K, N, kn, tid, ra, rbl, rbh);
            if (kn + BK < K) {
                __builtin_prefetch(A + (size_t)(m0 + (tid >> 3)) * K + kn + BK, 0, 1);
                __builtin_prefetch(B + (size_t)(kn + BK + (tid >> 5)) * N + n0, 0, 1);
            }
        }
        compute_chunk(As, Bs, wm, wn, half, l16, acc);
        __syncthreads();
    }

    // ---- epilogue: C/D layout -> M = r + 8*half, N = l16 ----
    #pragma unroll
    for (int tm = 0; tm < 2; ++tm)
        #pragma unroll
        for (int tn = 0; tn < 4; ++tn)
            #pragma unroll
            for (int r = 0; r < 8; ++r) {
                int gm = m0 + wm * 32 + tm * 16 + half * 8 + r;
                int gn = n0 + wn * 64 + tn * 16 + l16;
                C[(size_t)gm * N + gn] = acc[tm][tn][r];
            }
}

// ---------------------------------------------------------------------------
// Kernel 2: gate pre-activations g[t][0..127] = b + x[t] @ wih^T
// Hoisted out of the recurrence. grid = (T, 2 directions), block = 128.
// ---------------------------------------------------------------------------
__global__ void __launch_bounds__(128)
proj_kernel(const float* __restrict__ x, int ldx, int K,
            const float* __restrict__ wF, const float* __restrict__ bF,
            const float* __restrict__ wR, const float* __restrict__ bR,
            float* __restrict__ gF, float* __restrict__ gR) {
    const int t = blockIdx.x;
    const bool rev = (blockIdx.y != 0);
    const float* W  = rev ? wR : wF;
    const float* Bv = rev ? bR : bF;
    float* G = rev ? gR : gF;

    __shared__ float xs[1024];
    for (int k = threadIdx.x; k < K; k += 128) xs[k] = x[(size_t)t * ldx + k];
    __syncthreads();

    const int g = threadIdx.x;           // gate row 0..127
    const float4* wv = (const float4*)(W + (size_t)g * K);  // K % 4 == 0
    const float4* xv = (const float4*)xs;
    float acc = Bv[g];
    const int K4 = K >> 2;
    for (int q = 0; q < K4; ++q) {
        float4 a = wv[q], b = xv[q];
        acc = fmaf(a.x, b.x, acc);
        acc = fmaf(a.y, b.y, acc);
        acc = fmaf(a.z, b.z, acc);
        acc = fmaf(a.w, b.w, acc);
    }
    G[(size_t)t * 128 + g] = acc;
}

// ---------------------------------------------------------------------------
// Kernel 3: sequential LSTM scan, H = 32. One 128-thread block per direction
// (grid.x = 2 runs fwd + rev concurrently). Thread g owns gate row g:
// 32 whh weights live in VGPRs, h is broadcast through LDS.
// Gate order (torch): [0,32)=i [32,64)=f [64,96)=g [96,128)=o.
// ---------------------------------------------------------------------------
__global__ void __launch_bounds__(128)
lstm_scan(const float* __restrict__ g0, const float* __restrict__ g1,
          const float* __restrict__ w0, const float* __restrict__ w1,
          float* __restrict__ out, int col0_0, int col0_1,
          int rev0, int rev1, int ns0, int ns1, int T) {
    const int d = blockIdx.x;
    const float* G = d ? g1 : g0;
    const float* W = d ? w1 : w0;
    const int col0 = d ? col0_1 : col0_0;
    const int rev  = d ? rev1 : rev0;
    const int ns   = d ? ns1 : ns0;

    __shared__ float hbuf[32];
    __shared__ float gbuf[128];

    const int tid = threadIdx.x;
    float w[32];
    #pragma unroll
    for (int k = 0; k < 32; ++k) w[k] = W[tid * 32 + k];

    if (tid < 32) hbuf[tid] = 0.0f;
    float c = 0.0f;
    __syncthreads();

    for (int s = 0; s < ns; ++s) {
        const int t = rev ? (T - 1 - s) : s;
        float acc = G[(size_t)t * 128 + tid];
        const float4* h4 = (const float4*)hbuf;
        #pragma unroll
        for (int q = 0; q < 8; ++q) {
            float4 hv = h4[q];
            acc = fmaf(w[4 * q + 0], hv.x, acc);
            acc = fmaf(w[4 * q + 1], hv.y, acc);
            acc = fmaf(w[4 * q + 2], hv.z, acc);
            acc = fmaf(w[4 * q + 3], hv.w, acc);
        }
        gbuf[tid] = acc;
        __syncthreads();
        if (tid < 32) {
            float gi = sigm(gbuf[tid]);
            float gf = sigm(gbuf[32 + tid]);
            float gg = tanhf(gbuf[64 + tid]);
            float go = sigm(gbuf[96 + tid]);
            c = gf * c + gi * gg;
            float h = go * tanhf(c);
            hbuf[tid] = h;
            out[(size_t)t * 64 + col0 + tid] = h;
        }
        __syncthreads();
    }
}

// ---------------------------------------------------------------------------
// Kernel 4: emit final row: out[0:64] = layer2_out[t=0][0:64]
// ---------------------------------------------------------------------------
__global__ void copy_row0(const float* __restrict__ src, float* __restrict__ dst) {
    int i = threadIdx.x;
    if (i < 64) dst[i] = src[i];
}

// ---------------------------------------------------------------------------
// Host-side orchestration
// ---------------------------------------------------------------------------
extern "C" void kernel_launch(void* const* d_in, const int* in_sizes, int n_in,
                              void* d_out, int out_size, void* d_ws, size_t ws_size,
                              hipStream_t stream) {
    (void)in_sizes; (void)n_in; (void)out_size; (void)ws_size;

    const int T = 2048, V = 50264, E = 1024;

    const float* X     = (const float*)d_in[0];   // [T, V]
    const float* embed = (const float*)d_in[1];   // [V, E]
    // layer 0 (in_dim = 1024)
    const float* wih_f0 = (const float*)d_in[2];  // [128,1024]
    const float* whh_f0 = (const float*)d_in[3];  // [128,32]
    const float* b_f0   = (const float*)d_in[4];  // [128]
    const float* wih_r0 = (const float*)d_in[5];
    const float* whh_r0 = (const float*)d_in[6];
    const float* b_r0   = (const float*)d_in[7];
    // layer 1 (in_dim = 64)
    const float* wih_f1 = (const float*)d_in[8];  // [128,64]
    const float* whh_f1 = (const float*)d_in[9];
    const float* b_f1   = (const float*)d_in[10];
    const float* wih_r1 = (const float*)d_in[11];
    const float* whh_r1 = (const float*)d_in[12];
    const float* b_r1   = (const float*)d_in[13];

    float* ws = (float*)d_ws;
    float* Ebuf  = ws;                      size_t off = (size_t)T * E;      // [T,1024]
    float* G0f   = ws + off; off += (size_t)T * 128;                        // [T,128]
    float* G0r   = ws + off; off += (size_t)T * 128;
    float* L0out = ws + off; off += (size_t)T * 64;                         // [T,64]
    float* G1f   = ws + off; off += (size_t)T * 128;
    float* G1r   = ws + off; off += (size_t)T * 128;
    float* L1out = ws + off; off += (size_t)T * 64;

    // 1) Embedding GEMM: Ebuf = X @ embed   (bf16 WMMA, f32 accumulate)
    {
        dim3 grid(T / BM, E / BN);          // 16 x 8
        gemm_bf16_wmma<<<grid, 256, 0, stream>>>(X, embed, Ebuf, T, E, V);
    }

    // 2) Layer-0 input projections (both directions in one launch)
    proj_kernel<<<dim3(T, 2), 128, 0, stream>>>(Ebuf, E, E,
                                                wih_f0, b_f0, wih_r0, b_r0,
                                                G0f, G0r);

    // 3) Layer-0 scans: fwd -> cols [0,32), rev -> cols [32,64)
    lstm_scan<<<2, 128, 0, stream>>>(G0f, G0r, whh_f0, whh_r0, L0out,
                                     0, 32, 0, 1, T, T, T);

    // 4) Layer-1 input projections (K = 64)
    proj_kernel<<<dim3(T, 2), 128, 0, stream>>>(L0out, 64, 64,
                                                wih_f1, b_f1, wih_r1, b_r1,
                                                G1f, G1r);

    // 5) Layer-1 scans: forward output only needed at t=0 -> 1 step;
    //    reverse needs the full sequence to reach t=0.
    lstm_scan<<<2, 128, 0, stream>>>(G1f, G1r, whh_f1, whh_r1, L1out,
                                     0, 32, 0, 1, /*ns fwd*/ 1, /*ns rev*/ T, T);

    // 6) Final output = row 0 of layer-1 (second BiLSTM) output, [1,64]
    copy_row0<<<1, 64, 0, stream>>>(L1out, (float*)d_out);
}
